// ClassificationBackbone_29652454212159
// MI455X (gfx1250) — matrix operand order repaired and checked by reading, not compiled, not run on hardware
//
#include <hip/hip_runtime.h>
#include <hip/hip_bf16.h>
#include <stdint.h>

// ---------------------------------------------------------------------------
// Types for WMMA operands (wave32, gfx1250)
// ---------------------------------------------------------------------------
typedef __attribute__((ext_vector_type(16))) _Float16 v16h;
typedef __attribute__((ext_vector_type(8)))  float    v8f;
typedef __attribute__((ext_vector_type(2)))  float    v2f;

#define SELU_LAMBDA 1.0507009873554805f
#define SELU_ALPHA  1.6732632423543772f

__device__ __forceinline__ float selu(float x) {
    return x > 0.0f ? SELU_LAMBDA * x
                    : SELU_LAMBDA * SELU_ALPHA * (__expf(x) - 1.0f);
}

// ---------------------------------------------------------------------------
// Stage 0: h0[0:8] = (x/255)^0.25 ; h0[8:16] = x      (concat along batch)
// ---------------------------------------------------------------------------
__global__ void pow_concat_kernel(const float* __restrict__ x,
                                  float* __restrict__ h0, int n) {
    int i = blockIdx.x * blockDim.x + threadIdx.x;
    if (i >= n) return;
    float v = x[i];
    h0[i]     = sqrtf(sqrtf(v * (1.0f / 255.0f)));   // (v^.25)/255^.25
    h0[i + n] = v;
}

// ---------------------------------------------------------------------------
// Direct 3x3 VALID conv (fp32), optional bias, optional fused SELU.
// Small ragged channel counts (3..50) -> direct VALU; data lives in L2.
// ---------------------------------------------------------------------------
__global__ void conv3x3_kernel(const float* __restrict__ in,
                               const float* __restrict__ w,
                               const float* __restrict__ bias,
                               float* __restrict__ out,
                               int Bn, int IC, int OC, int H, int W, int doSelu) {
    const int OH = H - 2, OW = W - 2;
    long long total = (long long)Bn * OC * OH * OW;
    long long idx = (long long)blockIdx.x * blockDim.x + threadIdx.x;
    if (idx >= total) return;
    int ox = (int)(idx % OW);
    long long t = idx / OW;
    int oy = (int)(t % OH); t /= OH;
    int oc = (int)(t % OC);
    int b  = (int)(t / OC);

    float acc = bias ? bias[oc] : 0.0f;
    const float* wp = w + (long long)oc * IC * 9;
    const float* ip = in + (long long)b * IC * H * W;
    for (int ic = 0; ic < IC; ++ic) {
        const float* row = ip + (long long)ic * H * W + (long long)oy * W + ox;
        const float* wk  = wp + ic * 9;
        #pragma unroll
        for (int ky = 0; ky < 3; ++ky) {
            #pragma unroll
            for (int kx = 0; kx < 3; ++kx)
                acc += row[ky * W + kx] * wk[ky * 3 + kx];
        }
    }
    if (doSelu) acc = selu(acc);
    out[idx] = acc;
}

// ---------------------------------------------------------------------------
// 2x2 average pool (truncating odd trailing row/col, as the reference does)
// ---------------------------------------------------------------------------
__global__ void avgpool2_kernel(const float* __restrict__ in,
                                float* __restrict__ out,
                                int Bn, int C, int H, int W) {
    const int H2 = H / 2, W2 = W / 2;
    long long total = (long long)Bn * C * H2 * W2;
    long long idx = (long long)blockIdx.x * blockDim.x + threadIdx.x;
    if (idx >= total) return;
    int ox = (int)(idx % W2);
    long long t = idx / W2;
    int oy = (int)(t % H2); t /= H2;
    long long bc = t;  // b*C + c
    const float* p = in + (bc * H + 2 * oy) * W + 2 * ox;
    out[idx] = 0.25f * (p[0] + p[1] + p[W] + p[W + 1]);
}

// ---------------------------------------------------------------------------
// q/k/v projection (1x1 conv). q,k zero-padded to 8 channels (f32, so the
// chained 16x16x4 f32 WMMAs need no K masking); v padded to Cpad rows /
// Npad cols and stored f16 for the 16x16x32 f16 WMMA A operand.
// ---------------------------------------------------------------------------
__global__ void qkv_kernel(const float* __restrict__ x,
                           const float* __restrict__ wq, const float* __restrict__ bq,
                           const float* __restrict__ wk, const float* __restrict__ bk,
                           const float* __restrict__ wv, const float* __restrict__ bv,
                           float* __restrict__ qo, float* __restrict__ ko,
                           _Float16* __restrict__ vo,
                           int Bn, int C, int Cq, int N, int Npad, int Cpad) {
    int idx = blockIdx.x * blockDim.x + threadIdx.x;
    if (idx >= Bn * Npad) return;
    int n = idx % Npad;
    int b = idx / Npad;
    bool valid = (n < N);

    float xv[64];
    for (int c = 0; c < C; ++c)
        xv[c] = valid ? x[((long long)b * C + c) * N + n] : 0.0f;

    for (int r = 0; r < 8; ++r) {
        float qv = 0.0f, kv = 0.0f;
        if (r < Cq && valid) {
            for (int c = 0; c < C; ++c) {
                qv += wq[r * C + c] * xv[c];
                kv += wk[r * C + c] * xv[c];
            }
            qv += bq[r];
            kv += bk[r];
        }
        qo[((long long)b * 8 + r) * Npad + n] = qv;
        ko[((long long)b * 8 + r) * Npad + n] = kv;
    }
    for (int r = 0; r < Cpad; ++r) {
        float vv = 0.0f;
        if (r < C && valid) {
            for (int c = 0; c < C; ++c) vv += wv[r * C + c] * xv[c];
            vv += bv[r];
        }
        vo[((long long)b * Cpad + r) * Npad + n] = (_Float16)vv;
    }
}

// ---------------------------------------------------------------------------
// Flash-style self-attention tile kernel. One wave per (batch, 16-row n-tile).
//   energy  : V_WMMA_F32_16X16X4_F32  (two chained K=4 steps, Cq padded to 8)
//   P @ V   : V_WMMA_F32_16X16X32_F16 (NTC accumulator tiles over channels)
// Streaming softmax: the N x N energy matrix is never materialized; a 16x32
// tile lives in LDS. Accumulator rescale is per-lane (D lanes = n columns).
// out = gamma * (acc / rowsum) + x_in
// ---------------------------------------------------------------------------
template <int NTC>
__global__ __launch_bounds__(32)
void attn_kernel(const float* __restrict__ q, const float* __restrict__ k,
                 const _Float16* __restrict__ v, const float* __restrict__ xin,
                 float* __restrict__ out, const float* __restrict__ gamma,
                 int C, int N, int Npad, int nTiles) {
    __shared__ float lE[16][32];
    __shared__ float lrow[16];

    const int lane = threadIdx.x;
    const int half = lane >> 4;
    const int lr   = lane & 15;
    const int b  = blockIdx.x / nTiles;
    const int t  = blockIdx.x % nTiles;
    const int n0 = t * 16;

    const float* qb = q + (long long)b * 8 * Npad;
    const float* kb = k + (long long)b * 8 * Npad;

    // A operand for energy (q^T rows, loop invariant).
    // f32 16x4 A layout: element (M = lr, K = j + 2*half) in a[j].
    v2f aLo, aHi;
    aLo[0] = qb[(0 + 2 * half) * Npad + n0 + lr];
    aLo[1] = qb[(1 + 2 * half) * Npad + n0 + lr];
    aHi[0] = qb[(4 + 2 * half) * Npad + n0 + lr];
    aHi[1] = qb[(5 + 2 * half) * Npad + n0 + lr];

    v8f acc[NTC];
    #pragma unroll
    for (int tc = 0; tc < NTC; ++tc)
        acc[tc] = (v8f){0.f, 0.f, 0.f, 0.f, 0.f, 0.f, 0.f, 0.f};

    float mrun = -1e30f;
    float lsum = 0.0f;

    for (int m0 = 0; m0 < Npad; m0 += 32) {
        // ---- energy tiles E[16n x 32m] via f32 WMMA -----------------------
        v2f bLo, bHi;
        // f32 4x16 B layout: element (K = j + 2*half, N = lr) in b[j].
        bLo[0] = kb[(0 + 2 * half) * Npad + m0 + lr];
        bLo[1] = kb[(1 + 2 * half) * Npad + m0 + lr];
        bHi[0] = kb[(4 + 2 * half) * Npad + m0 + lr];
        bHi[1] = kb[(5 + 2 * half) * Npad + m0 + lr];
        v8f e0 = (v8f){0.f, 0.f, 0.f, 0.f, 0.f, 0.f, 0.f, 0.f};
        e0 = __builtin_amdgcn_wmma_f32_16x16x4_f32(false, aLo, false, bLo,
                                                   (short)0, e0, false, false);
        e0 = __builtin_amdgcn_wmma_f32_16x16x4_f32(false, aHi, false, bHi,
                                                   (short)0, e0, false, false);

        bLo[0] = kb[(0 + 2 * half) * Npad + m0 + 16 + lr];
        bLo[1] = kb[(1 + 2 * half) * Npad + m0 + 16 + lr];
        bHi[0] = kb[(4 + 2 * half) * Npad + m0 + 16 + lr];
        bHi[1] = kb[(5 + 2 * half) * Npad + m0 + 16 + lr];
        v8f e1 = (v8f){0.f, 0.f, 0.f, 0.f, 0.f, 0.f, 0.f, 0.f};
        e1 = __builtin_amdgcn_wmma_f32_16x16x4_f32(false, aLo, false, bLo,
                                                   (short)0, e1, false, false);
        e1 = __builtin_amdgcn_wmma_f32_16x16x4_f32(false, aHi, false, bHi,
                                                   (short)0, e1, false, false);

        // Stage to LDS: D element (M = j + 8*half, N = lr).
        #pragma unroll
        for (int j = 0; j < 8; ++j) {
            lE[j + 8 * half][lr]      = e0[j];
            lE[j + 8 * half][16 + lr] = e1[j];
        }
        __syncthreads();

        // ---- online softmax bookkeeping (lanes 0..15 own row = lane) ------
        if (lane < 16) {
            float tmax = mrun;
            for (int m = 0; m < 32; ++m)
                if (m0 + m < N) tmax = fmaxf(tmax, lE[lane][m]);
            float sc = __expf(mrun - tmax);
            float s  = 0.0f;
            for (int m = 0; m < 32; ++m) {
                float p = (m0 + m < N) ? __expf(lE[lane][m] - tmax) : 0.0f;
                lE[lane][m] = p;
                s += p;
            }
            lsum = lsum * sc + s;
            mrun = tmax;
            lrow[lane] = sc;
        }
        __syncthreads();

        // Rescale accumulators: D lanes = n columns, so per-lane multiply.
        float sc = lrow[lr];

        // Build P as f16 B operand: element (K = 8*half + 2i{+1} (+8 for i>=4), N = lr).
        v16h bP;
        #pragma unroll
        for (int i = 0; i < 8; ++i) {
            int mm = 8 * half + 2 * i + (i >= 4 ? 8 : 0);
            bP[2 * i]     = (_Float16)lE[lr][mm];
            bP[2 * i + 1] = (_Float16)lE[lr][mm + 1];
        }

        #pragma unroll
        for (int tc = 0; tc < NTC; ++tc) {
            #pragma unroll
            for (int j = 0; j < 8; ++j) acc[tc][j] *= sc;

            // V chunk as f16 A operand: row M = tc*16 + lr, K = m offset.
            const _Float16* vrow =
                v + ((long long)b * (NTC * 16) + tc * 16 + lr) * Npad + m0;
            v16h aV;
            #pragma unroll
            for (int i = 0; i < 4; ++i) {
                int k0 = half * 8 + 2 * i;
                int k1 = 16 + half * 8 + 2 * i;
                aV[2 * i]         = vrow[k0];
                aV[2 * i + 1]     = vrow[k0 + 1];
                aV[8 + 2 * i]     = vrow[k1];
                aV[8 + 2 * i + 1] = vrow[k1 + 1];
            }
            acc[tc] = __builtin_amdgcn_wmma_f32_16x16x32_f16(
                false, aV, false, bP, (short)0, acc[tc], false, false);
        }
        __syncthreads();  // lE reused next chunk
    }

    if (lane < 16) lrow[lane] = lsum;
    __syncthreads();
    const float denom = lrow[lr];
    const float g = gamma[0];
    const int n = n0 + lr;
    #pragma unroll
    for (int tc = 0; tc < NTC; ++tc) {
        #pragma unroll
        for (int j = 0; j < 8; ++j) {
            int c = tc * 16 + j + 8 * half;
            if (c < C && n < N) {
                long long o = ((long long)b * C + c) * N + n;
                out[o] = g * (acc[tc][j] / denom) + xin[o];
            }
        }
    }
}

// ---------------------------------------------------------------------------
// BatchNorm statistics per channel (mean over B,H,W; biased var) -> invstd
// ---------------------------------------------------------------------------
__global__ void bn_stats_kernel(const float* __restrict__ x,
                                float* __restrict__ mean,
                                float* __restrict__ invstd,
                                int Bn, int C, int HW) {
    __shared__ float s1[256];
    __shared__ float s2[256];
    const int c = blockIdx.x;
    const int tid = threadIdx.x;
    const int cnt = Bn * HW;
    float a = 0.0f, sq = 0.0f;
    for (int i = tid; i < cnt; i += blockDim.x) {
        int bb = i / HW, r = i % HW;
        float v = x[((long long)bb * C + c) * HW + r];
        a += v;
        sq += v * v;
    }
    s1[tid] = a; s2[tid] = sq;
    __syncthreads();
    for (int st = 128; st > 0; st >>= 1) {
        if (tid < st) { s1[tid] += s1[tid + st]; s2[tid] += s2[tid + st]; }
        __syncthreads();
    }
    if (tid == 0) {
        float m = s1[0] / cnt;
        float var = s2[0] / cnt - m * m;
        mean[c] = m;
        invstd[c] = rsqrtf(var + 1e-5f);
    }
}

__global__ void bn_apply_selu_kernel(const float* __restrict__ in,
                                     float* __restrict__ out,
                                     const float* __restrict__ mean,
                                     const float* __restrict__ invstd,
                                     const float* __restrict__ g,
                                     const float* __restrict__ beta,
                                     int Bn, int C, int HW) {
    long long total = (long long)Bn * C * HW;
    long long idx = (long long)blockIdx.x * blockDim.x + threadIdx.x;
    if (idx >= total) return;
    int c = (int)((idx / HW) % C);
    float y = (in[idx] - mean[c]) * invstd[c] * g[c] + beta[c];
    out[idx] = selu(y);
}

__global__ void copy_kernel(const float* __restrict__ src,
                            float* __restrict__ dst, int n) {
    int i = blockIdx.x * blockDim.x + threadIdx.x;
    if (i < n) dst[i] = src[i];
}

// ---------------------------------------------------------------------------
// Host orchestration
// ---------------------------------------------------------------------------
static inline int nblk(long long n, int b) { return (int)((n + b - 1) / b); }

extern "C" void kernel_launch(void* const* d_in, const int* in_sizes, int n_in,
                              void* d_out, int out_size, void* d_ws, size_t ws_size,
                              hipStream_t stream) {
    (void)in_sizes; (void)n_in; (void)out_size; (void)ws_size;
    const float* x      = (const float*)d_in[0];
    const float* enc_w1 = (const float*)d_in[1];
    const float* enc_b1 = (const float*)d_in[2];
    const float* enc_w2 = (const float*)d_in[3];
    const float* enc_b2 = (const float*)d_in[4];
    const float* a1_wq  = (const float*)d_in[5];
    const float* a1_bq  = (const float*)d_in[6];
    const float* a1_wk  = (const float*)d_in[7];
    const float* a1_bk  = (const float*)d_in[8];
    const float* a1_wv  = (const float*)d_in[9];
    const float* a1_bv  = (const float*)d_in[10];
    const float* a1_g   = (const float*)d_in[11];
    const float* c1_w   = (const float*)d_in[12];
    const float* c1_b   = (const float*)d_in[13];
    const float* a2_wq  = (const float*)d_in[14];
    const float* a2_bq  = (const float*)d_in[15];
    const float* a2_wk  = (const float*)d_in[16];
    const float* a2_bk  = (const float*)d_in[17];
    const float* a2_wv  = (const float*)d_in[18];
    const float* a2_bv  = (const float*)d_in[19];
    const float* a2_g   = (const float*)d_in[20];
    const float* c2_w   = (const float*)d_in[21];
    const float* c2_b   = (const float*)d_in[22];
    const float* c3_w   = (const float*)d_in[23];
    const float* c3_g   = (const float*)d_in[24];
    const float* c3_be  = (const float*)d_in[25];
    const float* c4_w   = (const float*)d_in[26];
    const float* c4_g   = (const float*)d_in[27];
    const float* c4_be  = (const float*)d_in[28];
    const float* a3_wq  = (const float*)d_in[29];
    const float* a3_bq  = (const float*)d_in[30];
    const float* a3_wk  = (const float*)d_in[31];
    const float* a3_bk  = (const float*)d_in[32];
    const float* a3_wv  = (const float*)d_in[33];
    const float* a3_bv  = (const float*)d_in[34];
    const float* a3_g   = (const float*)d_in[35];

    uint8_t* w8 = (uint8_t*)d_ws;
    // Three reused regions (total ~71 MB):
    const size_t R0 = 0;            // 50,331,648 B
    const size_t R1 = 50331648;     //  8,388,608 B
    const size_t R2 = 58720256;     // 12,582,912 B
    #define F32AT(off) ((float*)(w8 + (off)))
    #define F16AT(off) ((_Float16*)(w8 + (off)))

    float* fout = (float*)d_out;
    const int BLK = 256;

    // ---- encoder (doubled batch of 16) ----------------------------------
    const int nX = 8 * 3 * 200 * 200;
    float* h0 = F32AT(R1);
    pow_concat_kernel<<<nblk(nX, BLK), BLK, 0, stream>>>(x, h0, nX);

    float* c1o = F32AT(R0);   // (16,20,198,198)
    conv3x3_kernel<<<nblk(16LL * 20 * 198 * 198, BLK), BLK, 0, stream>>>(
        h0, enc_w1, enc_b1, c1o, 16, 3, 20, 200, 200, 1);

    float* p1 = F32AT(R2);    // (16,20,99,99)
    avgpool2_kernel<<<nblk(16LL * 20 * 99 * 99, BLK), BLK, 0, stream>>>(
        c1o, p1, 16, 20, 198, 198);

    float* c2o = F32AT(R0);   // (16,40,97,97)
    conv3x3_kernel<<<nblk(16LL * 40 * 97 * 97, BLK), BLK, 0, stream>>>(
        p1, enc_w2, enc_b2, c2o, 16, 20, 40, 99, 99, 1);

    float* henc = F32AT(R1);  // (16,40,48,48)
    avgpool2_kernel<<<nblk(16LL * 40 * 48 * 48, BLK), BLK, 0, stream>>>(
        c2o, henc, 16, 40, 97, 97);

    // x_enc output (batches 8..15)
    const int nEnc = 8 * 40 * 48 * 48;  // 737280
    copy_kernel<<<nblk(nEnc, BLK), BLK, 0, stream>>>(henc + nEnc, fout + 5120, nEnc);

    // ---- attention 1: C=40, Cq=5, N=2304 --------------------------------
    {
        const int C = 40, Cq = 5, N = 2304, Npad = 2304, Cpad = 48, nT = 144;
        float*    q1 = F32AT(R2 + 0);
        float*    k1 = F32AT(R2 + 589824);
        _Float16* v1 = F16AT(R2 + 1179648);
        float*    o1 = F32AT(R2 + 2949120);
        qkv_kernel<<<nblk(8LL * Npad, BLK), BLK, 0, stream>>>(
            henc, a1_wq, a1_bq, a1_wk, a1_bk, a1_wv, a1_bv,
            q1, k1, v1, 8, C, Cq, N, Npad, Cpad);
        attn_kernel<3><<<8 * nT, 32, 0, stream>>>(q1, k1, v1, henc, o1, a1_g,
                                                  C, N, Npad, nT);
        // c1 conv + selu: (8,50,46,46)
        conv3x3_kernel<<<nblk(8LL * 50 * 46 * 46, BLK), BLK, 0, stream>>>(
            o1, c1_w, c1_b, F32AT(R0), 8, 40, 50, 48, 48, 1);
    }

    // ---- attention 2: C=50, Cq=6, N=2116 --------------------------------
    {
        const int C = 50, Cq = 6, N = 2116, Npad = 2144, Cpad = 64, nT = 133;
        float* xin = F32AT(R0);
        float*    q2 = F32AT(R1 + 0);
        float*    k2 = F32AT(R1 + 548864);
        _Float16* v2 = F16AT(R1 + 1097728);
        float*    o2 = F32AT(R2);
        qkv_kernel<<<nblk(8LL * Npad, BLK), BLK, 0, stream>>>(
            xin, a2_wq, a2_bq, a2_wk, a2_bk, a2_wv, a2_bv,
            q2, k2, v2, 8, C, Cq, N, Npad, Cpad);
        attn_kernel<4><<<8 * nT, 32, 0, stream>>>(q2, k2, v2, xin, o2, a2_g,
                                                  C, N, Npad, nT);
        // c2 conv + selu: (8,40,44,44) -> pool (8,40,22,22)
        conv3x3_kernel<<<nblk(8LL * 40 * 44 * 44, BLK), BLK, 0, stream>>>(
            o2, c2_w, c2_b, F32AT(R0), 8, 50, 40, 46, 46, 1);
        avgpool2_kernel<<<nblk(8LL * 40 * 22 * 22, BLK), BLK, 0, stream>>>(
            F32AT(R0), F32AT(R1), 8, 40, 44, 44);
    }

    // ---- c3 conv (no bias) + BN + selu + pool ---------------------------
    conv3x3_kernel<<<nblk(8LL * 30 * 20 * 20, BLK), BLK, 0, stream>>>(
        F32AT(R1), c3_w, nullptr, F32AT(R0), 8, 40, 30, 22, 22, 0);
    bn_stats_kernel<<<30, 256, 0, stream>>>(F32AT(R0), F32AT(R2),
                                            F32AT(R2 + 4096), 8, 30, 400);
    bn_apply_selu_kernel<<<nblk(8LL * 30 * 400, BLK), BLK, 0, stream>>>(
        F32AT(R0), F32AT(R1), F32AT(R2), F32AT(R2 + 4096), c3_g, c3_be, 8, 30, 400);
    avgpool2_kernel<<<nblk(8LL * 30 * 10 * 10, BLK), BLK, 0, stream>>>(
        F32AT(R1), F32AT(R0), 8, 30, 20, 20);

    // ---- c4 conv (no bias) + BN + selu ----------------------------------
    conv3x3_kernel<<<nblk(8LL * 10 * 8 * 8, BLK), BLK, 0, stream>>>(
        F32AT(R0), c4_w, nullptr, F32AT(R1), 8, 30, 10, 10, 10, 0);
    bn_stats_kernel<<<10, 256, 0, stream>>>(F32AT(R1), F32AT(R2),
                                            F32AT(R2 + 4096), 8, 10, 64);
    bn_apply_selu_kernel<<<nblk(8LL * 10 * 64, BLK), BLK, 0, stream>>>(
        F32AT(R1), F32AT(R0), F32AT(R2), F32AT(R2 + 4096), c4_g, c4_be, 8, 10, 64);

    // ---- attention 3: C=10, Cq=1, N=64 -> writes d_out[0:5120] ----------
    {
        const int C = 10, Cq = 1, N = 64, Npad = 64, Cpad = 16, nT = 4;
        float* xin = F32AT(R0);
        float*    q3 = F32AT(R2 + 65536);
        float*    k3 = F32AT(R2 + 65536 + 16384);
        _Float16* v3 = F16AT(R2 + 65536 + 32768);
        qkv_kernel<<<nblk(8LL * Npad, BLK), BLK, 0, stream>>>(
            xin, a3_wq, a3_bq, a3_wk, a3_bk, a3_wv, a3_bv,
            q3, k3, v3, 8, C, Cq, N, Npad, Cpad);
        attn_kernel<1><<<8 * nT, 32, 0, stream>>>(q3, k3, v3, xin, fout, a3_g,
                                                  C, N, Npad, nT);
    }
    #undef F32AT
    #undef F16AT
}
